// LSTM_Seq2Seq_70239895159204
// MI455X (gfx1250) — matrix operand order repaired and checked by reading, not compiled, not run on hardware
//
#include <hip/hip_runtime.h>
#include <hip/hip_bf16.h>

// ---------------------------------------------------------------------------
// Problem constants (match reference)
// ---------------------------------------------------------------------------
#define HID    1024
#define DIN    80
#define TLEN   512
#define BATCH  64
#define KP     128              // x-features padded 80 -> 128
#define KTOT   (HID + KP)       // 1152 concatenated K  (h | x)
#define KSTEPS (KTOT / 32)      // 36 WMMA k-steps
#define NWG_DIR 16              // workgroups per direction (64 hidden cols each)
#define NWG     (2 * NWG_DIR)   // total persistent workgroups

// LDS layout (dynamic):
//   [0, 148480)            : A staging  64 rows x 2320B (1160 bf16, 8-elem pad)
//                            -- after the k-loop this region is reused as the
//                               gate exchange buffer G[4][64][66] f32 (67584B)
//   [148480, 165376)       : persistent cell state C[64][66] f32
#define A_PITCH_B  2320
#define G_PITCH    66
#define LDS_A_BYTES (64 * A_PITCH_B)
#define LDS_BYTES   (LDS_A_BYTES + 64 * G_PITCH * 4)   // 165376

// ---------------------------------------------------------------------------
// Types
// ---------------------------------------------------------------------------
typedef __attribute__((ext_vector_type(16))) __bf16 v16bf;
typedef __attribute__((ext_vector_type(8)))  float  v8f;

union BF16x16 { uint4 u[2]; v16bf v; };

__device__ __forceinline__ __bf16 f2bf(float f) {
    union { float f; unsigned u; } v; v.f = f;
    unsigned r = v.u + 0x7FFFu + ((v.u >> 16) & 1u);   // round-to-nearest-even
    unsigned short s = (unsigned short)(r >> 16);
    __bf16 out; __builtin_memcpy(&out, &s, 2); return out;
}

__device__ __forceinline__ float fast_sigmoid(float x) {
    return 1.0f / (1.0f + __expf(-x));
}
__device__ __forceinline__ float fast_tanh(float x) {
    float e = __expf(2.0f * x);
    return (e - 1.0f) / (e + 1.0f);
}

// ---- CDNA5 async global->LDS copy (ASYNCcnt-tracked, no VGPR data) --------
__device__ __forceinline__ void async_g2l_b128(const void* gsrc, void* ldst) {
    unsigned lds_off = (unsigned)(size_t)ldst;                   // LDS byte addr
    unsigned long long ga = (unsigned long long)(size_t)gsrc;    // 64-bit global
    asm volatile("global_load_async_to_lds_b128 %0, %1, off"
                 :: "v"(lds_off), "v"(ga) : "memory");
}

__device__ __forceinline__ void wait_asynccnt0() {
#if defined(__has_builtin)
#if __has_builtin(__builtin_amdgcn_s_wait_asynccnt)
    __builtin_amdgcn_s_wait_asynccnt(0);
    return;
#else
    asm volatile("s_wait_asynccnt 0x0" ::: "memory");
    return;
#endif
#else
    asm volatile("s_wait_asynccnt 0x0" ::: "memory");
#endif
}

// Device-wide barrier (all NWG blocks resident: 32 WGs, 1 per WGP)
__device__ __forceinline__ void gridBarrier(unsigned* cnt, unsigned* gen) {
    __syncthreads();
    if (threadIdx.x == 0) {
        __threadfence();
        unsigned g = __hip_atomic_load(gen, __ATOMIC_RELAXED, __HIP_MEMORY_SCOPE_AGENT);
        unsigned arrived = __hip_atomic_fetch_add(cnt, 1u, __ATOMIC_ACQ_REL, __HIP_MEMORY_SCOPE_AGENT);
        if (arrived == (unsigned)(NWG - 1)) {
            __hip_atomic_store(cnt, 0u, __ATOMIC_RELAXED, __HIP_MEMORY_SCOPE_AGENT);
            __hip_atomic_fetch_add(gen, 1u, __ATOMIC_ACQ_REL, __HIP_MEMORY_SCOPE_AGENT);
        } else {
            while (__hip_atomic_load(gen, __ATOMIC_ACQUIRE, __HIP_MEMORY_SCOPE_AGENT) == g) {
                __builtin_amdgcn_s_sleep(2);
            }
        }
        __threadfence();
    }
    __syncthreads();
}

// ---------------------------------------------------------------------------
// Prep kernels (all parallel, run once per call)
// ---------------------------------------------------------------------------
__global__ void prep_zero(unsigned* bar, unsigned* hbuf_u32, int n) {
    int i = blockIdx.x * blockDim.x + threadIdx.x;
    if (i < 2) bar[i] = 0u;
    if (i < n) hbuf_u32[i] = 0u;
}

// x0 = input[-1] @ W_in.T + b_in  ->  xs[0]  (bf16, padded to KP)
__global__ void prep_x0(const float* __restrict__ input,
                        const float* __restrict__ W_in,
                        const float* __restrict__ b_in,
                        __bf16* __restrict__ xs) {
    int i = blockIdx.x * blockDim.x + threadIdx.x;
    if (i >= BATCH * KP) return;
    int b = i >> 7, d = i & (KP - 1);
    float acc = 0.0f;
    if (d < DIN) {
        const float* ip = input + (size_t)(TLEN - 1) * BATCH * HID + (size_t)b * HID;
        const float* wp = W_in + (size_t)d * HID;
        for (int k = 0; k < HID; ++k) acc = fmaf(ip[k], wp[k], acc);
        acc += b_in[d];
    }
    xs[i] = f2bf((d < DIN) ? acc : 0.0f);
}

// xs[1..T-1] = target[0..T-2]  (bf16, padded)
__global__ void prep_xs(const float* __restrict__ target, __bf16* __restrict__ xs) {
    int i = blockIdx.x * blockDim.x + threadIdx.x;
    if (i >= (TLEN - 1) * BATCH * KP) return;
    int t1 = i / (BATCH * KP);                 // 0..510 -> step t1+1
    int r  = i - t1 * (BATCH * KP);
    int b  = r >> 7, d = r & (KP - 1);
    float v = (d < DIN) ? target[((size_t)t1 * BATCH + b) * DIN + d] : 0.0f;
    xs[(size_t)(t1 + 1) * BATCH * KP + r] = f2bf(v);
}

// Wcat[dir][n][k] = k<1024 ? W_hh[n][k] : (k<1104 ? W_ih[n][k-1024] : 0)   (bf16)
__global__ void prep_wcat(const float* __restrict__ fW_hh, const float* __restrict__ fW_ih,
                          const float* __restrict__ bW_hh, const float* __restrict__ bW_ih,
                          __bf16* __restrict__ Wcat) {
    int i = blockIdx.x * blockDim.x + threadIdx.x;
    if (i >= 2 * 4 * HID * KTOT) return;
    int dir = i / (4 * HID * KTOT);
    int r   = i - dir * (4 * HID * KTOT);
    int n   = r / KTOT;
    int k   = r - n * KTOT;
    const float* Whh = dir ? bW_hh : fW_hh;
    const float* Wih = dir ? bW_ih : fW_ih;
    float v = 0.0f;
    if (k < HID)            v = Whh[(size_t)n * HID + k];
    else if (k < HID + DIN) v = Wih[(size_t)n * DIN + (k - HID)];
    Wcat[i] = f2bf(v);
}

// Wout[dir][n][k] = W_out[n][k]  (bf16, row-major == B-operand layout)
__global__ void prep_wout(const float* __restrict__ fW_out, const float* __restrict__ bW_out,
                          __bf16* __restrict__ Wout) {
    int i = blockIdx.x * blockDim.x + threadIdx.x;
    if (i >= 2 * DIN * HID) return;
    int dir = i / (DIN * HID);
    int r   = i - dir * (DIN * HID);
    Wout[i] = f2bf(dir ? bW_out[r] : fW_out[r]);
}

// ---------------------------------------------------------------------------
// Phase 2: persistent recurrent kernel.
//   grid = 32 WGs x 256 threads. WG (dir, w) owns hidden cols [w*64, w*64+64).
//   Wave v (0..7): gate = v>>1, j-sub = (v&1)*32 -> 2 N-tiles, 4 M-tiles.
// ---------------------------------------------------------------------------
__global__ void lstm_step_kernel(const __bf16* __restrict__ xs,     // [T][B][KP]
                                 const __bf16* __restrict__ Wcat,   // [2][4H][KTOT]
                                 const float*  __restrict__ fb,     // [4H]
                                 const float*  __restrict__ bb,     // [4H]
                                 __bf16* __restrict__ h_buf,        // [2][2][B][H]
                                 __bf16* __restrict__ h_hist,       // [2][T][B][H]
                                 unsigned* __restrict__ bar) {
    extern __shared__ char smem[];
    char*  A_lds = smem;                                   // staging, dead after k-loop
    float* G_lds = (float*)smem;                           // aliased gate buffer [4][64][66]
    float* C_lds = (float*)(smem + LDS_A_BYTES);           // persistent cell state [64][66]

    const int tid  = threadIdx.x;
    const int wave = tid >> 5;
    const int lane = tid & 31;
    const int wg   = blockIdx.x;
    const int dir  = wg >> 4;
    const int w    = wg & 15;

    const float*  bgate = dir ? bb : fb;
    const __bf16* Wd    = Wcat + (size_t)dir * 4 * HID * KTOT;
    __bf16* hb = h_buf + (size_t)dir * 2 * BATCH * HID;
    __bf16* hh = h_hist + (size_t)dir * TLEN * BATCH * HID;

    // init cell state
    for (int i = tid; i < 64 * G_PITCH; i += 256) C_lds[i] = 0.0f;

    const int gate  = wave >> 1;               // 0..3 (i,f,g,o)
    const int jsub  = (wave & 1) * 32;         // 0 or 32
    const int ncol0 = gate * HID + w * 64 + jsub;
    const int half  = lane >> 4;               // K half select
    const int mlane = lane & 15;               // row (A) / column (B,C)

    const char* brow0 = (const char*)(Wd + (size_t)(ncol0 + mlane) * KTOT);
    const char* brow1 = (const char*)(Wd + (size_t)(ncol0 + 16 + mlane) * KTOT);

    for (int t = 0; t < TLEN; ++t) {
        const int cur = t & 1, nxt = cur ^ 1;

        // ---- 1. stage A = [h_cur | x_t] into LDS (64 x 1152 bf16) ----
        //         via CDNA5 async global->LDS DMA (ASYNCcnt tracked)
        const __bf16* hcur = hb + (size_t)cur * BATCH * HID;
        const __bf16* xrow = xs + (size_t)t * BATCH * KP;
        for (int idx = tid; idx < 64 * 144; idx += 256) {
            int row = idx / 144;
            int q   = idx - row * 144;
            const uint4* src = (q < 128)
                ? ((const uint4*)(hcur + (size_t)row * HID)) + q
                : ((const uint4*)(xrow + (size_t)row * KP)) + (q - 128);
            async_g2l_b128(src, A_lds + row * A_PITCH_B + q * 16);
        }
        wait_asynccnt0();
        __syncthreads();

        // ---- 2. WMMA k-loop: gates += A @ Wcat.T (bf16 in, f32 accum) ----
        v8f zero = {0.f,0.f,0.f,0.f,0.f,0.f,0.f,0.f};
        v8f acc[4][2];
        #pragma unroll
        for (int m = 0; m < 4; ++m) { acc[m][0] = zero; acc[m][1] = zero; }

        for (int ks = 0; ks < KSTEPS; ++ks) {
            const int k0 = ks * 32;
            BF16x16 a[4];
            #pragma unroll
            for (int m = 0; m < 4; ++m) {
                const char* ar = A_lds + (m * 16 + mlane) * A_PITCH_B;
                a[m].u[0] = *(const uint4*)(ar + (k0 + half * 16) * 2);
                a[m].u[1] = *(const uint4*)(ar + (k0 + half * 16 + 8) * 2);
            }
            BF16x16 bm[2];
            {
                const char* p0 = brow0 + (k0 + half * 16) * 2;
                const char* p1 = brow1 + (k0 + half * 16) * 2;
                bm[0].u[0] = *(const uint4*)(p0);
                bm[0].u[1] = *(const uint4*)(p0 + 16);
                bm[1].u[0] = *(const uint4*)(p1);
                bm[1].u[1] = *(const uint4*)(p1 + 16);
            }
            #pragma unroll
            for (int m = 0; m < 4; ++m) {
                #pragma unroll
                for (int nt = 0; nt < 2; ++nt) {
                    acc[m][nt] = __builtin_amdgcn_wmma_f32_16x16x32_bf16(
                        false, a[m].v, false, bm[nt].v, (short)0, acc[m][nt],
                        false, false);
                }
            }
        }
        __syncthreads();   // A region dead -> reuse as G

        // ---- 3. scatter gates (+bias) to LDS exchange buffer ----
        #pragma unroll
        for (int nt = 0; nt < 2; ++nt) {
            const int jj = jsub + nt * 16 + mlane;            // 0..63
            const float bv = bgate[gate * HID + w * 64 + jj];
            #pragma unroll
            for (int m = 0; m < 4; ++m) {
                const int rb = m * 16 + half * 8;
                #pragma unroll
                for (int r = 0; r < 8; ++r) {
                    G_lds[(gate * 64 + rb + r) * G_PITCH + jj] = acc[m][nt][r] + bv;
                }
            }
        }
        __syncthreads();

        // ---- 4. LSTM elementwise update, emit h (bf16) ----
        for (int i = tid; i < BATCH * 64; i += 256) {
            const int b = i >> 6, j = i & 63;
            const float gi = G_lds[(0 * 64 + b) * G_PITCH + j];
            const float gf = G_lds[(1 * 64 + b) * G_PITCH + j];
            const float gg = G_lds[(2 * 64 + b) * G_PITCH + j];
            const float go = G_lds[(3 * 64 + b) * G_PITCH + j];
            const float cc = fast_sigmoid(gf) * C_lds[b * G_PITCH + j]
                           + fast_sigmoid(gi) * fast_tanh(gg);
            C_lds[b * G_PITCH + j] = cc;
            const float hv = fast_sigmoid(go) * fast_tanh(cc);
            const __bf16 h16 = f2bf(hv);
            const int col = w * 64 + j;
            hb[(size_t)nxt * BATCH * HID + (size_t)b * HID + col] = h16;
            hh[(size_t)t * BATCH * HID + (size_t)b * HID + col]  = h16;
        }

        // ---- 5. device-wide step barrier ----
        gridBarrier(bar, bar + 1);
    }
}

// ---------------------------------------------------------------------------
// Phase 3: y = h_hist @ W_out.T + b_out   -> out [2][T][B][80]  (f32)
//   One 16x16 tile per wave; M = T*B = 32768 (2048 tiles), N = 80 (5 tiles).
// ---------------------------------------------------------------------------
__global__ void out_gemm(const __bf16* __restrict__ h_hist,  // [2][T*B][H]
                         const __bf16* __restrict__ Wout,    // [2][80][H]
                         const float*  __restrict__ fb_out,
                         const float*  __restrict__ bb_out,
                         float* __restrict__ out) {
    const int wave = threadIdx.x >> 5;
    const int lane = threadIdx.x & 31;
    const int gid  = blockIdx.x * 8 + wave;        // 0..20479
    const int dir  = gid / (2048 * 5);
    const int rem  = gid - dir * (2048 * 5);
    const int mt   = rem / 5;
    const int nt   = rem - mt * 5;

    const int half  = lane >> 4;
    const int mlane = lane & 15;

    const char* arow = (const char*)(h_hist + (size_t)dir * TLEN * BATCH * HID
                                            + (size_t)(mt * 16 + mlane) * HID);
    const char* brow = (const char*)(Wout + (size_t)dir * DIN * HID
                                          + (size_t)(nt * 16 + mlane) * HID);
    v8f acc = {0.f,0.f,0.f,0.f,0.f,0.f,0.f,0.f};
    for (int ks = 0; ks < HID / 32; ++ks) {
        const int k0 = ks * 32;
        BF16x16 a, b;
        a.u[0] = *(const uint4*)(arow + (k0 + half * 16) * 2);
        a.u[1] = *(const uint4*)(arow + (k0 + half * 16 + 8) * 2);
        const char* bp = brow + (k0 + half * 16) * 2;
        b.u[0] = *(const uint4*)(bp);
        b.u[1] = *(const uint4*)(bp + 16);
        acc = __builtin_amdgcn_wmma_f32_16x16x32_bf16(
            false, a.v, false, b.v, (short)0, acc, false, false);
    }

    const int col   = nt * 16 + mlane;             // < 80
    const int Mbase = mt * 16 + half * 8;
    const float bo  = (dir ? bb_out : fb_out)[col];
    float* op = out + (size_t)dir * TLEN * BATCH * DIN + (size_t)Mbase * DIN + col;
    #pragma unroll
    for (int r = 0; r < 8; ++r) op[(size_t)r * DIN] = acc[r] + bo;
}

// ---------------------------------------------------------------------------
// Host launcher
// ---------------------------------------------------------------------------
extern "C" void kernel_launch(void* const* d_in, const int* in_sizes, int n_in,
                              void* d_out, int out_size, void* d_ws, size_t ws_size,
                              hipStream_t stream) {
    (void)in_sizes; (void)n_in; (void)out_size; (void)ws_size;

    const float* input   = (const float*)d_in[0];
    const float* target  = (const float*)d_in[1];
    const float* W_in    = (const float*)d_in[2];
    const float* b_in    = (const float*)d_in[3];
    const float* fW_ih   = (const float*)d_in[4];
    const float* fW_hh   = (const float*)d_in[5];
    const float* fb      = (const float*)d_in[6];
    const float* fW_out  = (const float*)d_in[7];
    const float* fb_out  = (const float*)d_in[8];
    const float* bW_ih   = (const float*)d_in[9];
    const float* bW_hh   = (const float*)d_in[10];
    const float* bb      = (const float*)d_in[11];
    const float* bW_out  = (const float*)d_in[12];
    const float* bb_out  = (const float*)d_in[13];
    float* out = (float*)d_out;

    // workspace carve-up (all offsets 256B aligned)
    char* ws = (char*)d_ws;
    size_t o = 0;
    unsigned* bar = (unsigned*)(ws + o);            o += 256;
    __bf16* xs     = (__bf16*)(ws + o);             o += (size_t)TLEN * BATCH * KP * 2;       // 8 MB
    __bf16* Wcat   = (__bf16*)(ws + o);             o += (size_t)2 * 4 * HID * KTOT * 2;      // 18 MB
    __bf16* Wout   = (__bf16*)(ws + o);             o += (size_t)2 * DIN * HID * 2;           // 320 KB
    __bf16* h_buf  = (__bf16*)(ws + o);             o += (size_t)2 * 2 * BATCH * HID * 2;     // 512 KB
    __bf16* h_hist = (__bf16*)(ws + o);             o += (size_t)2 * TLEN * BATCH * HID * 2;  // 128 MB

    // ---- prep (parallel) ----
    {
        int n = (2 * 2 * BATCH * HID * 2) / 4;      // h_buf as u32
        prep_zero<<<(n + 255) / 256, 256, 0, stream>>>(bar, (unsigned*)h_buf, n);
    }
    prep_x0 <<<(BATCH * KP + 255) / 256, 256, 0, stream>>>(input, W_in, b_in, xs);
    prep_xs <<<((TLEN - 1) * BATCH * KP + 255) / 256, 256, 0, stream>>>(target, xs);
    prep_wcat<<<(2 * 4 * HID * KTOT + 255) / 256, 256, 0, stream>>>(fW_hh, fW_ih, bW_hh, bW_ih, Wcat);
    prep_wout<<<(2 * DIN * HID + 255) / 256, 256, 0, stream>>>(fW_out, bW_out, Wout);

    // ---- phase 2: persistent recurrence (32 WGs, 1/WGP, big dynamic LDS) ----
    hipFuncSetAttribute((const void*)lstm_step_kernel,
                        hipFuncAttributeMaxDynamicSharedMemorySize, LDS_BYTES);
    lstm_step_kernel<<<NWG, 256, LDS_BYTES, stream>>>(xs, Wcat, fb, bb, h_buf, h_hist, bar);

    // ---- phase 3: output projection GEMM ----
    out_gemm<<<(2 * 2048 * 5) / 8, 256, 0, stream>>>(h_hist, Wout, fb_out, bb_out, out);
}
